// M3GNet_49984829390863
// MI455X (gfx1250) — compile-verified, matrix operand order checked by reading, not run on hardware
//
#include <hip/hip_runtime.h>
#include <math.h>

#define DN 128
#define DEG 9
#define DINX 384
#define CUTOFF_R 5.0f
#define TB_CUTOFF_R 4.0f
#define SBF_FACTOR 0.12649110640673517f
#define Y00C 0.28209479177387814f
#define Y10C 0.4886025119029199f
#define Y20C 0.31539156525252005f

typedef __attribute__((ext_vector_type(16))) _Float16 v16h;
typedef __attribute__((ext_vector_type(8)))  float    v8f;

union V16U { v16h h; uint4 u[2]; };

__device__ __constant__ float c_jroots[3][3] = {
    {3.141592653589793f, 6.283185307179586f, 9.42477796076938f},
    {4.493409457909064f, 7.725251836937707f, 10.904121659428899f},
    {5.763459196894550f, 9.095011330476355f, 12.322940970566582f}};

__device__ inline float sigm(float x) { return 1.0f / (1.0f + __expf(-x)); }
__device__ inline float silu(float x) { return x * sigm(x); }

__device__ inline float jlf(int l, float x) {
    float s = __sinf(x), c = __cosf(x);
    float x2 = x * x;
    if (l == 0) return s / x;
    if (l == 1) return s / x2 - c / x;
    if (l == 2) return (3.0f / (x2 * x) - 1.0f / x) * s - 3.0f * c / x2;
    return (15.0f / (x2 * x2) - 6.0f / x2) * s - (15.0f / (x2 * x) - 1.0f / x) * c;
}

// ---------------------------------------------------------------- small kernels

__global__ void k_zero(float* p, int n) {
    int i = blockIdx.x * blockDim.x + threadIdx.x;
    if (i < n) p[i] = 0.0f;
}

__global__ void k_addinto(float* a, const float* b, int n) {
    int i = blockIdx.x * blockDim.x + threadIdx.x;
    if (i < n) a[i] += b[i];
}

// convert fp32 row-major [K x Ncols] -> f16 column-major [Ncols x K]
__global__ void k_w16(const float* src, _Float16* dst, int K, int Ncols) {
    int i = blockIdx.x * blockDim.x + threadIdx.x;
    if (i >= K * Ncols) return;
    int n = i / K, k = i - n * K;
    dst[i] = (_Float16)src[(size_t)k * Ncols + n];
}

__global__ void k_nodeinit(const float* emb, const int* nt, float* nf, int N) {
    int i = blockIdx.x * blockDim.x + threadIdx.x;
    if (i >= N * DN) return;
    int n = i >> 7, c = i & 127;
    nf[i] = emb[(size_t)nt[n] * DN + c];
}

__global__ void k_edge_geom(const float* pos, const int* src, const int* dst,
                            float* bvec, float* bdist, float* rbf, float* cut3, int E) {
    int e = blockIdx.x * blockDim.x + threadIdx.x;
    if (e >= E) return;
    int s = src[e], d = dst[e];
    float vx = pos[d * 3 + 0] - pos[s * 3 + 0];
    float vy = pos[d * 3 + 1] - pos[s * 3 + 1];
    float vz = pos[d * 3 + 2] - pos[s * 3 + 2];
    float r = sqrtf(vx * vx + vy * vy + vz * vz + 1e-12f);
    bvec[e * 3 + 0] = vx; bvec[e * 3 + 1] = vy; bvec[e * 3 + 2] = vz;
    bdist[e] = r;
#pragma unroll
    for (int l = 0; l < 3; ++l)
#pragma unroll
        for (int n = 0; n < 3; ++n) {
            float root = c_jroots[l][n];
            float norm = fabsf(jlf(l + 1, root));
            rbf[(size_t)e * DEG + l * 3 + n] = jlf(l, r * root / CUTOFF_R) * (SBF_FACTOR / norm);
        }
    float x = r * (1.0f / TB_CUTOFF_R);
    float x2 = x * x, x3 = x2 * x;
    cut3[e] = 1.0f - 6.0f * x3 * x2 + 15.0f * x2 * x2 - 10.0f * x3;
}

__global__ void k_edgeemb(const float* rbf, const float* We, const float* be, float* ef, int E) {
    int i = blockIdx.x * blockDim.x + threadIdx.x;
    if (i >= E * DN) return;
    int e = i >> 7, c = i & 127;
    float s = be[c];
#pragma unroll
    for (int q = 0; q < DEG; ++q) s += rbf[(size_t)e * DEG + q] * We[q * DN + c];
    ef[i] = silu(s);
}

__global__ void k_triplet(const float* bvec, const float* bdist, const float* rbf,
                          const float* cut3, const int* t_src, const int* t_dst,
                          const int* dst, float* tb9, int* eat, int T) {
    int t = blockIdx.x * blockDim.x + threadIdx.x;
    if (t >= T) return;
    int a = t_src[t], b = t_dst[t];
    float dotv = bvec[a * 3 + 0] * bvec[b * 3 + 0] + bvec[a * 3 + 1] * bvec[b * 3 + 1] +
                 bvec[a * 3 + 2] * bvec[b * 3 + 2];
    float ct = dotv / (bdist[a] * bdist[b]);
    float sh[3] = {Y00C, Y10C * ct, Y20C * (3.0f * ct * ct - 1.0f)};
    float tw = cut3[a] * cut3[b];
    const float* rb = rbf + (size_t)b * DEG;
#pragma unroll
    for (int l = 0; l < 3; ++l)
#pragma unroll
        for (int n = 0; n < 3; ++n)
            tb9[(size_t)t * DEG + l * 3 + n] = rb[l * 3 + n] * sh[l] * tw;
    eat[t] = dst[b];
}

__global__ void k_atoms(const float* nf, const float* Wa, const float* ba, float* atoms, int N) {
    int i = blockIdx.x * blockDim.x + threadIdx.x;
    if (i >= N * DEG) return;
    int n = i / DEG, j = i - n * DEG;
    float s = ba[j];
    for (int d = 0; d < DN; ++d) s += nf[(size_t)n * DN + d] * Wa[d * DEG + j];
    atoms[i] = sigm(s);
}

__global__ void k_scatter3(const float* tb9, const int* eat, const int* t_src,
                           const float* atoms, float* nbuf, int T) {
    int t = blockIdx.x * blockDim.x + threadIdx.x;
    if (t >= T) return;
    int e = t_src[t];
    const float* a = atoms + (size_t)eat[t] * DEG;
    const float* b = tb9 + (size_t)t * DEG;
#pragma unroll
    for (int j = 0; j < DEG; ++j) atomicAdd(&nbuf[(size_t)e * DEG + j], b[j] * a[j]);
}

__global__ void k_tbedge(const float* nbuf, const float* Wb, const float* Wg, float* ef, int E) {
    int i = blockIdx.x * blockDim.x + threadIdx.x;
    if (i >= E * DN) return;
    int e = i >> 7, c = i & 127;
    const float* nb = nbuf + (size_t)e * DEG;
    float s = 0.0f, sg = 0.0f;
#pragma unroll
    for (int q = 0; q < DEG; ++q) {
        s += nb[q] * Wb[q * DN + c];
        sg += nb[q] * Wg[q * DN + c];
    }
    ef[i] += silu(s) * sigm(sg);
}

// ---------------------------------------------------------------- WMMA gated MLP

#define ROWS 32   // rows (edges) per wave = 2 M-tiles

// A fragment: 16-row tile from LDS (row-major, stride sh halves), K block kb.
__device__ inline v16h ldA(const _Float16* xs, int sh, int kb, int lane) {
    int m = lane & 15, hi = (lane >> 4) & 1;
    const _Float16* p = xs + m * sh + kb + hi * 8;
    V16U f;
    f.u[0] = *(const uint4*)(p);
    f.u[1] = *(const uint4*)(p + 16);
    return f.h;
}

// B fragment: weights are f16 column-major [128 x K]; tile cols nb..nb+15.
__device__ inline v16h ldB(const _Float16* w, int K, int nb, int kb, int lane) {
    int n = nb + (lane & 15), hi = (lane >> 4) & 1;
    const _Float16* p = w + (size_t)n * K + kb + hi * 8;
    V16U f;
    f.u[0] = *(const uint4*)(p);
    f.u[1] = *(const uint4*)(p + 16);
    return f.h;
}

// 2 M-tiles x 8 N-tiles; B fragment reused across both M tiles.
__device__ inline void gemm2x8(const _Float16* a, int sh, const _Float16* w, int K,
                               v8f acc[2][8], int lane) {
    for (int kb = 0; kb < K; kb += 32) {
        v16h a0 = ldA(a, sh, kb, lane);
        v16h a1 = ldA(a + 16 * sh, sh, kb, lane);
        if (kb + 32 < K)  // prefetch next K-block of the weight stream
            __builtin_prefetch((const void*)(w + (size_t)(lane & 15) * K + kb + 32), 0, 1);
#pragma unroll
        for (int t = 0; t < 8; ++t) {
            v16h bf = ldB(w, K, t * 16, kb, lane);
            acc[0][t] = __builtin_amdgcn_wmma_f32_16x16x32_f16(
                false, a0, false, bf, (short)0, acc[0][t], false, false);
            acc[1][t] = __builtin_amdgcn_wmma_f32_16x16x32_f16(
                false, a1, false, bf, (short)0, acc[1][t], false, false);
        }
    }
}

__device__ inline void store_silu(_Float16* out, const v8f acc[2][8], const float* bias, int lane) {
    int nlo = lane & 15, hi = (lane >> 4) & 1;
#pragma unroll
    for (int mi = 0; mi < 2; ++mi)
#pragma unroll
        for (int t = 0; t < 8; ++t)
#pragma unroll
            for (int j = 0; j < 8; ++j) {
                int m = mi * 16 + j + hi * 8, n = t * 16 + nlo;
                out[m * DN + n] = (_Float16)silu(acc[mi][t][j] + bias[n]);
            }
}

__global__ __launch_bounds__(32)
void k_gconv(const float* nf, const int* src, const int* dst, float* ef, const float* rbf,
             const _Float16* hW0, const _Float16* hW1, const _Float16* hW2,
             const _Float16* gW0, const _Float16* gW1, const _Float16* gW2,
             const float* b0, const float* b1, const float* b2,
             const float* gb0, const float* gb1, const float* gb2,
             const float* wt, float* nacc, int mode, int E) {
    __shared__ __align__(16) _Float16 xs[ROWS * DINX];
    __shared__ __align__(16) _Float16 ta[ROWS * DN];
    __shared__ __align__(16) _Float16 tb[ROWS * DN];
    __shared__ __align__(16) _Float16 hout[ROWS * DN];
    __shared__ __align__(16) _Float16 wfs[ROWS * DN];
    __shared__ int sI[ROWS], dI[ROWS];
    int lane = threadIdx.x;
    int m0 = blockIdx.x * ROWS;
    {
        int e = m0 + lane; if (e >= E) e = E - 1;
        sI[lane] = src[e]; dI[lane] = dst[e];
    }
    __syncthreads();
    // stage concat(nf[src], nf[dst], ef) as f16
    for (int idx = lane; idx < ROWS * DINX; idx += 32) {
        int r = idx / DINX, c = idx - r * DINX;
        int e = m0 + r; if (e >= E) e = E - 1;
        float v;
        if (c < DN)            v = nf[(size_t)sI[r] * DN + c];
        else if (c < 2 * DN)   v = nf[(size_t)dI[r] * DN + (c - DN)];
        else                   v = ef[(size_t)e * DN + (c - 2 * DN)];
        xs[idx] = (_Float16)v;
    }
    // rbf @ wt scaling factors
    for (int idx = lane; idx < ROWS * DN; idx += 32) {
        int r = idx >> 7, n = idx & 127;
        int e = m0 + r; if (e >= E) e = E - 1;
        const float* rb = rbf + (size_t)e * DEG;
        float s = 0.0f;
#pragma unroll
        for (int q = 0; q < DEG; ++q) s += rb[q] * wt[q * DN + n];
        wfs[idx] = (_Float16)s;
    }
    __syncthreads();
    int nlo = lane & 15, hi = (lane >> 4) & 1;
    // ---- value chain: silu(silu(silu(x@W0)@W1)@W2) ----
    { v8f a[2][8] = {}; gemm2x8(xs, DINX, hW0, DINX, a, lane); store_silu(ta, a, b0, lane); }
    __syncthreads();
    { v8f a[2][8] = {}; gemm2x8(ta, DN, hW1, DN, a, lane); store_silu(tb, a, b1, lane); }
    __syncthreads();
    {
        v8f a[2][8] = {};
        gemm2x8(tb, DN, hW2, DN, a, lane);
        store_silu(hout, a, b2, lane);  // final silu(h) parked in LDS
    }
    __syncthreads();
    // ---- gate chain: sigmoid(silu(silu(x@gW0)@gW1)@gW2) ----
    { v8f a[2][8] = {}; gemm2x8(xs, DINX, gW0, DINX, a, lane); store_silu(ta, a, gb0, lane); }
    __syncthreads();
    { v8f a[2][8] = {}; gemm2x8(ta, DN, gW1, DN, a, lane); store_silu(tb, a, gb1, lane); }
    __syncthreads();
    v8f gf[2][8] = {};
    gemm2x8(tb, DN, gW2, DN, gf, lane);
    // ---- combine + writeback ----
#pragma unroll
    for (int mi = 0; mi < 2; ++mi)
#pragma unroll
        for (int t = 0; t < 8; ++t)
#pragma unroll
            for (int j = 0; j < 8; ++j) {
                int m = mi * 16 + j + hi * 8, n = t * 16 + nlo;
                int e = m0 + m;
                if (e >= E) continue;
                float g = sigm(gf[mi][t][j] + gb2[n]);
                float o = (float)hout[m * DN + n] * g * (float)wfs[m * DN + n];
                if (mode == 0) ef[(size_t)e * DN + n] += o;
                else atomicAdd(&nacc[(size_t)dI[m] * DN + n], o);
            }
}

// ---------------------------------------------------------------- readout

__global__ void k_colmean(const float* nf, float* v, int N) {
    __shared__ float sbuf[256];
    int c = blockIdx.x;
    float s = 0.0f;
    for (int n = threadIdx.x; n < N; n += 256) s += nf[(size_t)n * DN + c];
    sbuf[threadIdx.x] = s;
    __syncthreads();
    for (int off = 128; off; off >>= 1) {
        if ((int)threadIdx.x < off) sbuf[threadIdx.x] += sbuf[threadIdx.x + off];
        __syncthreads();
    }
    if (threadIdx.x == 0) v[c] = sbuf[0] / (float)N;
}

__global__ void k_final(const float* vmean, const float* fW0, const float* fb0,
                        const float* fW1, const float* fb1, const float* fW2,
                        const float* fb2, float* out) {
    __shared__ float v[DN], h[DN], r[DN];
    int t = threadIdx.x;
    v[t] = vmean[t];
    __syncthreads();
    float s = fb0[t];
    for (int d = 0; d < DN; ++d) s += v[d] * fW0[d * DN + t];
    h[t] = silu(s);
    __syncthreads();
    s = fb1[t];
    for (int d = 0; d < DN; ++d) s += h[d] * fW1[d * DN + t];
    r[t] = silu(s) * fW2[t];
    __syncthreads();
    for (int off = 64; off; off >>= 1) {
        if (t < off) r[t] += r[t + off];
        __syncthreads();
    }
    if (t == 0) out[0] = r[0] + fb2[0];
}

// ---------------------------------------------------------------- launch

extern "C" void kernel_launch(void* const* d_in, const int* in_sizes, int n_in,
                              void* d_out, int out_size, void* d_ws, size_t ws_size,
                              hipStream_t stream) {
    const float* pos       = (const float*)d_in[0];
    const int*   node_type = (const int*)d_in[1];
    const int*   src       = (const int*)d_in[2];
    const int*   dst       = (const int*)d_in[3];
    const int*   t_src     = (const int*)d_in[4];
    const int*   t_dst     = (const int*)d_in[5];
    const float* emb       = (const float*)d_in[6];
    const float* eW        = (const float*)d_in[7];
    const float* eb        = (const float*)d_in[8];
    const float* tbaW      = (const float*)d_in[9];
    const float* tbab      = (const float*)d_in[10];
    const float* tbbW      = (const float*)d_in[11];
    const float* tbbWg     = (const float*)d_in[12];
    const float* W0        = (const float*)d_in[13];
    const float* b0        = (const float*)d_in[14];
    const float* W1        = (const float*)d_in[15];
    const float* b1        = (const float*)d_in[16];
    const float* W2        = (const float*)d_in[17];
    const float* b2        = (const float*)d_in[18];
    const float* gW0       = (const float*)d_in[19];
    const float* gb0       = (const float*)d_in[20];
    const float* gW1       = (const float*)d_in[21];
    const float* gb1       = (const float*)d_in[22];
    const float* gW2       = (const float*)d_in[23];
    const float* gb2       = (const float*)d_in[24];
    const float* wt        = (const float*)d_in[25];
    const float* fW0       = (const float*)d_in[26];
    const float* fb0       = (const float*)d_in[27];
    const float* fW1       = (const float*)d_in[28];
    const float* fb1       = (const float*)d_in[29];
    const float* fW2       = (const float*)d_in[30];
    const float* fb2       = (const float*)d_in[31];

    const int N = in_sizes[1];
    const int E = in_sizes[2];
    const int T = in_sizes[4];

    char* wsp = (char*)d_ws;
    auto alloc = [&](size_t bytes) -> void* {
        void* p = (void*)wsp;
        wsp += (bytes + 255) & ~(size_t)255;
        return p;
    };
    float*     ef    = (float*)alloc((size_t)E * DN * 4);
    float*     nf    = (float*)alloc((size_t)N * DN * 4);
    float*     nacc  = (float*)alloc((size_t)N * DN * 4);
    float*     rbf   = (float*)alloc((size_t)E * DEG * 4);
    float*     cut3  = (float*)alloc((size_t)E * 4);
    float*     bvec  = (float*)alloc((size_t)E * 3 * 4);
    float*     bdist = (float*)alloc((size_t)E * 4);
    float*     tb9   = (float*)alloc((size_t)T * DEG * 4);
    int*       eat   = (int*)alloc((size_t)T * 4);
    float*     atoms = (float*)alloc((size_t)N * DEG * 4);
    float*     nbuf  = (float*)alloc((size_t)E * DEG * 4);
    _Float16*  w16   = (_Float16*)alloc((size_t)6 * 163840 * 2);
    float*     vmean = (float*)alloc(DN * 4);

    const int TPB = 256;
    auto grid = [](long long n, int tpb) { return (unsigned)((n + tpb - 1) / tpb); };

    // per-call weight convert/transpose into f16 column-major
    for (int k = 0; k < 3; ++k)
        for (int br = 0; br < 2; ++br) {
            _Float16* base = w16 + (size_t)(k * 2 + br) * 163840;
            const size_t kb = (size_t)(k * 2 + br);
            k_w16<<<grid(DINX * DN, TPB), TPB, 0, stream>>>(W0 + kb * DINX * DN, base, DINX, DN);
            k_w16<<<grid(DN * DN, TPB), TPB, 0, stream>>>(W1 + kb * DN * DN, base + 49152, DN, DN);
            k_w16<<<grid(DN * DN, TPB), TPB, 0, stream>>>(W2 + kb * DN * DN, base + 65536, DN, DN);
            k_w16<<<grid(DINX * DN, TPB), TPB, 0, stream>>>(gW0 + kb * DINX * DN, base + 81920, DINX, DN);
            k_w16<<<grid(DN * DN, TPB), TPB, 0, stream>>>(gW1 + kb * DN * DN, base + 131072, DN, DN);
            k_w16<<<grid(DN * DN, TPB), TPB, 0, stream>>>(gW2 + kb * DN * DN, base + 147456, DN, DN);
        }

    k_edge_geom<<<grid(E, TPB), TPB, 0, stream>>>(pos, src, dst, bvec, bdist, rbf, cut3, E);
    k_edgeemb<<<grid((long long)E * DN, TPB), TPB, 0, stream>>>(rbf, eW, eb, ef, E);
    k_nodeinit<<<grid((long long)N * DN, TPB), TPB, 0, stream>>>(emb, node_type, nf, N);
    k_triplet<<<grid(T, TPB), TPB, 0, stream>>>(bvec, bdist, rbf, cut3, t_src, t_dst, dst, tb9, eat, T);

    for (int k = 0; k < 3; ++k) {
        k_atoms<<<grid((long long)N * DEG, TPB), TPB, 0, stream>>>(
            nf, tbaW + (size_t)k * DN * DEG, tbab + (size_t)k * DEG, atoms, N);
        k_zero<<<grid((long long)E * DEG, TPB), TPB, 0, stream>>>(nbuf, E * DEG);
        k_scatter3<<<grid(T, TPB), TPB, 0, stream>>>(tb9, eat, t_src, atoms, nbuf, T);
        k_tbedge<<<grid((long long)E * DN, TPB), TPB, 0, stream>>>(
            nbuf, tbbW + (size_t)k * DEG * DN, tbbWg + (size_t)k * DEG * DN, ef, E);

        // --- edge update (WMMA) ---
        {
            _Float16* wb = w16 + (size_t)(k * 2 + 0) * 163840;
            size_t o = (size_t)(k * 2 + 0);
            k_gconv<<<grid(E, ROWS), 32, 0, stream>>>(
                nf, src, dst, ef, rbf,
                wb, wb + 49152, wb + 65536, wb + 81920, wb + 131072, wb + 147456,
                b0 + o * DN, b1 + o * DN, b2 + o * DN,
                gb0 + o * DN, gb1 + o * DN, gb2 + o * DN,
                wt + o * DEG * DN, nacc, 0, E);
        }
        // --- node update (WMMA + scatter) ---
        k_zero<<<grid((long long)N * DN, TPB), TPB, 0, stream>>>(nacc, N * DN);
        {
            _Float16* wb = w16 + (size_t)(k * 2 + 1) * 163840;
            size_t o = (size_t)(k * 2 + 1);
            k_gconv<<<grid(E, ROWS), 32, 0, stream>>>(
                nf, src, dst, ef, rbf,
                wb, wb + 49152, wb + 65536, wb + 81920, wb + 131072, wb + 147456,
                b0 + o * DN, b1 + o * DN, b2 + o * DN,
                gb0 + o * DN, gb1 + o * DN, gb2 + o * DN,
                wt + o * DEG * DN, nacc, 1, E);
        }
        k_addinto<<<grid((long long)N * DN, TPB), TPB, 0, stream>>>(nf, nacc, N * DN);
    }

    k_colmean<<<DN, 256, 0, stream>>>(nf, vmean, N);
    k_final<<<1, DN, 0, stream>>>(vmean, fW0, fb0, fW1, fb1, fW2, fb2, (float*)d_out);
}
